// GCN_TCN_Unit_49057116455499
// MI455X (gfx1250) — matrix operand order repaired and checked by reading, not compiled, not run on hardware
//
#include <hip/hip_runtime.h>

// ---------------------------------------------------------------------------
// GCN+TCN unit for MI455X (gfx1250, wave32).
// Channel GEMMs run on v_wmma_f32_16x16x32_bf16 with f32 accumulation.
// Activations are kept channel-contiguous ("transposed", [n][p][c]) so every
// WMMA B fragment is one contiguous 32B load; weights are pre-swizzled into
// the A-fragment lane layout so A is also one contiguous 32B load.
// ---------------------------------------------------------------------------

#define Nn 32
#define Cc 128
#define Tt 256
#define Vv 25
#define Pp (Tt * Vv)          // 6400 positions per (n, c) plane
#define NC (Nn * Cc)          // 4096 instance-norm planes
#define EPSI 1e-5f

typedef __attribute__((ext_vector_type(16))) __bf16 v16bf;
typedef __attribute__((ext_vector_type(8)))  float  v8f;

__device__ __forceinline__ unsigned short f2bf(float f) {
  union { float f; unsigned u; } x; x.f = f;
  unsigned r = x.u + 0x7fffu + ((x.u >> 16) & 1u);   // round-to-nearest-even
  return (unsigned short)(r >> 16);
}
__device__ __forceinline__ float bf2f(unsigned short u) {
  union { unsigned u; float f; } x; x.u = ((unsigned)u) << 16; return x.f;
}

// sum across each 16-lane half via ds_swizzle XOR patterns (imm offsets)
__device__ __forceinline__ float half16_sum(float v) {
  union { float f; int i; } a, b;
  a.f = v;
  b.i = __builtin_amdgcn_ds_swizzle(a.i, 0x041f); a.f += b.f;  // xor 1
  b.i = __builtin_amdgcn_ds_swizzle(a.i, 0x081f); a.f += b.f;  // xor 2
  b.i = __builtin_amdgcn_ds_swizzle(a.i, 0x101f); a.f += b.f;  // xor 4
  b.i = __builtin_amdgcn_ds_swizzle(a.i, 0x201f); a.f += b.f;  // xor 8
  return a.f;
}

// ---------------------------------------------------------------------------
// 0) pre-swizzle Wd / Wt into WMMA A-fragment lane layout (bf16):
//    Wdsw[((ot*4+cc)*32 + lane)*16 + e]       = Wd[o0+lm, cc*32 + kA(e,half)]
//    Wtsw[(((k*8+ot)*4+cc)*32 + lane)*16 + e] = Wt[o0+lm, cc*32 + kA(e,half), k]
//    kA(e,half) = 16*half + e   (contiguous k order, matches B fragments)
// ---------------------------------------------------------------------------
__global__ __launch_bounds__(256) void prep_weights(
    const float* __restrict__ Wd, const float* __restrict__ Wt,
    unsigned short* __restrict__ Wdsw, unsigned short* __restrict__ Wtsw) {
  int i = blockIdx.x * 256 + threadIdx.x;              // 0 .. 81919
  int e    = i & 15;
  int lane = (i >> 4) & 31;
  int cc   = (i >> 9) & 3;
  int ot   = (i >> 11) & 7;
  int k    = i >> 14;
  int lm = lane & 15, half = lane >> 4;
  int c  = cc * 32 + 16 * half + e;
  Wtsw[i] = f2bf(Wt[(ot * 16 + lm) * (Cc * 5) + c * 5 + k]);
  if (i < 16384) Wdsw[i] = f2bf(Wd[(ot * 16 + lm) * Cc + c]);
}

// ---------------------------------------------------------------------------
// 1) gT[n,p,c] = sum_v x[n,c,t,v] * PA[v,w]   (p = t*25+w; K=25, VALU)
//    Output is channel-contiguous so GEMM-2 B fragments are contiguous.
// ---------------------------------------------------------------------------
__global__ __launch_bounds__(256) void gcn_kernel(
    const float* __restrict__ x, const float* __restrict__ PA,
    unsigned short* __restrict__ gT) {
  size_t idx = (size_t)blockIdx.x * 256 + threadIdx.x;   // over [n][p][c]
  int    c = (int)(idx & (Cc - 1));
  size_t q = idx >> 7;
  int    p = (int)(q % Pp);
  int    n = (int)(q / Pp);
  int    t = p / Vv, w = p - t * Vv;
  const float* xr = x + (((size_t)(n * Cc + c)) * Tt + t) * Vv;
  float s = 0.f;
#pragma unroll
  for (int v = 0; v < Vv; ++v) s += xr[v] * PA[v * Vv + w];   // PA col uniform
  gT[idx] = f2bf(s);
}

// ---------------------------------------------------------------------------
// 2) y_lin[o,p] = sum_c Wd[o,c] * g[c,p] per n, WMMA bf16 -> f32.
//    Wave = 16(o) x 16(p) tile; 4 unrolled K=32 chunks; fragments are single
//    contiguous 32B loads. D stored transposed+packed (one b128 per lane).
// ---------------------------------------------------------------------------
__global__ __launch_bounds__(128) void cgemm_wmma(
    const unsigned short* __restrict__ gT, const unsigned short* __restrict__ Wdsw,
    unsigned short* __restrict__ yT, float* __restrict__ stats) {
  const int lane = threadIdx.x & 31;
  const int wv   = threadIdx.x >> 5;
  const int half = lane >> 4;
  const int lm   = lane & 15;
  const int p    = (blockIdx.x * 4 + wv) * 16 + lm;
  const int oti  = blockIdx.y, o0 = oti * 16;
  const int n    = blockIdx.z;

  const unsigned short* gn = gT + (size_t)n * Pp * Cc + (size_t)p * Cc + 16 * half;
  const v16bf*          A  = ((const v16bf*)Wdsw) + (size_t)(oti * 4) * 32 + lane;

  v8f acc = {};
#pragma unroll
  for (int cc = 0; cc < 4; ++cc) {
    v16bf a = A[cc * 32];                          // 32B contiguous
    v16bf b = *(const v16bf*)(gn + cc * 32);       // 32B contiguous
    acc = __builtin_amdgcn_wmma_f32_16x16x32_bf16(
        false, a, false, b, (short)0, acc, false, false);
  }

  // D: per lane, rows o0+8*half .. +7 at column p are contiguous in yT
  uint4 pk;
  pk.x = (unsigned)f2bf(acc[0]) | ((unsigned)f2bf(acc[1]) << 16);
  pk.y = (unsigned)f2bf(acc[2]) | ((unsigned)f2bf(acc[3]) << 16);
  pk.z = (unsigned)f2bf(acc[4]) | ((unsigned)f2bf(acc[5]) << 16);
  pk.w = (unsigned)f2bf(acc[6]) | ((unsigned)f2bf(acc[7]) << 16);
  *(uint4*)(yT + (size_t)n * Pp * Cc + (size_t)p * Cc + o0 + 8 * half) = pk;

#pragma unroll
  for (int r = 0; r < 8; ++r) {                    // instance-norm partial stats
    float v  = acc[r];
    float sv = half16_sum(v);
    float sq = half16_sum(v * v);
    if (lm == 0) {
      int o = o0 + r + 8 * half;
      atomicAdd(&stats[n * Cc + o], sv);
      atomicAdd(&stats[NC + n * Cc + o], sq);
    }
  }
}

// ---------------------------------------------------------------------------
// 3) y = relu(instnorm(y_lin) + x), in place on yT (bf16).
//    bd cancels exactly under affine-free instance norm.
// ---------------------------------------------------------------------------
__global__ __launch_bounds__(256) void inorm_res_relu(
    unsigned short* __restrict__ yT, const float* __restrict__ x,
    const float* __restrict__ stats) {
  size_t idx = (size_t)blockIdx.x * 256 + threadIdx.x;   // over [n][p][c]
  int    c = (int)(idx & (Cc - 1));
  size_t q = idx >> 7;
  int    p = (int)(q % Pp);
  int    n = (int)(q / Pp);
  int    plane = n * Cc + c;
  float  m   = stats[plane] * (1.f / Pp);
  float  var = stats[NC + plane] * (1.f / Pp) - m * m;
  float  inv = rsqrtf(var + EPSI);
  float  v   = (bf2f(yT[idx]) - m) * inv + x[(size_t)plane * Pp + p];
  yT[idx]    = f2bf(fmaxf(v, 0.f));
}

// ---------------------------------------------------------------------------
// 4) z[o,p] = sum_k sum_c y[c, reflect(t,k)*25+v] * Wt[o,c,k], WMMA bf16.
//    Reflect padding folded into the per-lane B base pointer; 20 unrolled
//    WMMAs per tile, each fed by two contiguous 32B loads.
// ---------------------------------------------------------------------------
__global__ __launch_bounds__(128) void tcn_wmma(
    const unsigned short* __restrict__ yT, const unsigned short* __restrict__ Wtsw,
    unsigned short* __restrict__ z, float* __restrict__ stats) {
  const int lane = threadIdx.x & 31;
  const int wv   = threadIdx.x >> 5;
  const int half = lane >> 4;
  const int lm   = lane & 15;
  const int p    = (blockIdx.x * 4 + wv) * 16 + lm;
  const int oti  = blockIdx.y, o0 = oti * 16;
  const int n    = blockIdx.z;
  const int t    = p / Vv, v = p - t * Vv;

  const unsigned short* yn = yT + (size_t)n * Pp * Cc;
  v8f acc = {};
#pragma unroll
  for (int k = 0; k < 5; ++k) {
    int st = t - 4 + k;
    if (st < 0) st = -st;                          // reflect pad (top only)
    const unsigned short* bcol = yn + (size_t)(st * Vv + v) * Cc + 16 * half;
    const v16bf* A = ((const v16bf*)Wtsw) + (size_t)((k * 8 + oti) * 4) * 32 + lane;
#pragma unroll
    for (int cc = 0; cc < 4; ++cc) {
      v16bf a = A[cc * 32];
      v16bf b = *(const v16bf*)(bcol + cc * 32);
      acc = __builtin_amdgcn_wmma_f32_16x16x32_bf16(
          false, a, false, b, (short)0, acc, false, false);
    }
  }

  unsigned short* zn = z + (size_t)n * Cc * Pp;    // natural [n][o][p] for output
#pragma unroll
  for (int r = 0; r < 8; ++r) {
    int   o  = o0 + r + 8 * half;
    float vv = acc[r];
    zn[(size_t)o * Pp + p] = f2bf(vv);
    float sv = half16_sum(vv);
    float sq = half16_sum(vv * vv);
    if (lm == 0) {
      atomicAdd(&stats[n * Cc + o], sv);
      atomicAdd(&stats[NC + n * Cc + o], sq);
    }
  }
}

// ---------------------------------------------------------------------------
// 5) out = relu(instnorm(z) + x), f32, fully coalesced. bt cancels like bd.
// ---------------------------------------------------------------------------
__global__ __launch_bounds__(256) void final_norm_res(
    const unsigned short* __restrict__ zbuf, const float* __restrict__ x,
    const float* __restrict__ stats, float* __restrict__ out) {
  size_t idx   = (size_t)blockIdx.x * 256 + threadIdx.x;   // over [n][c][p]
  int    plane = (int)(idx / Pp);
  float  m   = stats[plane] * (1.f / Pp);
  float  var = stats[NC + plane] * (1.f / Pp) - m * m;
  float  inv = rsqrtf(var + EPSI);
  float  vv  = (bf2f(zbuf[idx]) - m) * inv + x[idx];
  out[idx]   = fmaxf(vv, 0.f);
}

// ---------------------------------------------------------------------------
extern "C" void kernel_launch(void* const* d_in, const int* in_sizes, int n_in,
                              void* d_out, int out_size, void* d_ws, size_t ws_size,
                              hipStream_t stream) {
  (void)in_sizes; (void)n_in; (void)out_size; (void)ws_size;

  const float* x  = (const float*)d_in[0];
  const float* PA = (const float*)d_in[1];
  const float* Wd = (const float*)d_in[2];
  // d_in[3]=bd, d_in[5]=bt: per-channel constants cancel exactly under
  // affine-free instance norm -> mathematically no-ops in this block.
  const float* Wt = (const float*)d_in[4];
  float* out = (float*)d_out;

  char* ws = (char*)d_ws;
  const size_t ACT_BF16 = (size_t)Nn * Cc * Pp * 2;                 // 52,428,800 B
  unsigned short* gT   = (unsigned short*)(ws);                     // gT, reused for z
  unsigned short* yT   = (unsigned short*)(ws + ACT_BF16);
  unsigned short* Wdsw = (unsigned short*)(ws + 2 * ACT_BF16);
  unsigned short* Wtsw = (unsigned short*)(ws + 2 * ACT_BF16 + 32768);
  float* stats1 = (float*)(ws + 2 * ACT_BF16 + 32768 + 163840);
  float* stats2 = (float*)(ws + 2 * ACT_BF16 + 32768 + 163840 + 32768);

  hipMemsetAsync(stats1, 0, 2 * 32768, stream);                     // both stat bufs

  const size_t total = (size_t)Nn * Cc * Pp;                        // 26,214,400

  prep_weights<<<(Cc * Cc * 5 + 255) / 256, 256, 0, stream>>>(Wd, Wt, Wdsw, Wtsw);
  gcn_kernel<<<(unsigned)(total / 256), 256, 0, stream>>>(x, PA, gT);

  dim3 gg(Pp / 16 / 4, Cc / 16, Nn);                                // (100, 8, 32)
  cgemm_wmma<<<gg, 128, 0, stream>>>(gT, Wdsw, yT, stats1);
  inorm_res_relu<<<(unsigned)(total / 256), 256, 0, stream>>>(yT, x, stats1);
  tcn_wmma<<<gg, 128, 0, stream>>>(yT, Wtsw, gT, stats2);           // z into gT space
  final_norm_res<<<(unsigned)(total / 256), 256, 0, stream>>>(gT, x, stats2, out);
}
